// LSTM_11965778886820
// MI455X (gfx1250) — matrix operand order repaired and checked by reading, not compiled, not run on hardware
//
#include <hip/hip_runtime.h>
#include <hip/hip_bf16.h>

// Problem dims
#define Bb   64
#define Tt   1024
#define DIN  256
#define Hh   512
#define NWG  32        // H/16: each WG owns 16 h-columns (i.e. 64 gate rows)
#define NTHR 256       // 8 waves of 32
#define KB0  24        // (H + DIN)/32  = layer0 K-blocks
#define KB1  32        // (2H)/32       = layer1 K-blocks

// LDS layout (bytes)
#define W0LDS   0
#define W0BYTES (4 * KB0 * 1024)          // 98304: 4 gate tile-cols x 24 kb x 1KB frag
#define W1LDS   (W0LDS + W0BYTES)
#define W1BYTES (4 * KB1 * 1024)          // 131072
#define GATESO  (W1LDS + W1BYTES)         // 229376: [2 layers][4 gates][64 rows][16 cols] f32
#define GATESB  (2 * 64 * 64 * 4)         // 32768
#define CSTO    (GATESO + GATESB)         // 262144: [2 layers][64][16] f32 cell state
#define CSTB    (2 * 1024 * 4)            // 8192
#define LDS_TOTAL (CSTO + CSTB)           // 270336  (< 320KB WGP LDS)

// Workspace layout (bytes)
#define H0OFF  0                          // bf16 [2 parity][B][H]
#define H1OFF  (2 * Bb * Hh * 2)          // 131072
#define BAROFF (2 * H1OFF)                // 262144: {count, generation}
#define WSWORDS ((BAROFF / 4) + 2)

// Output layout (floats)
#define OUT_HN ((size_t)Bb * Tt * Hh)         // outputs (B,T,H)
#define OUT_CN (OUT_HN + (size_t)2 * Bb * Hh) // h_n (2,B,H) then c_n (2,B,H)

typedef __attribute__((ext_vector_type(16))) __bf16       v16bf;
typedef __attribute__((ext_vector_type(8)))  float        v8f;
typedef __attribute__((ext_vector_type(4)))  unsigned int u32x4;

struct Pair128 { u32x4 lo, hi; };
union FragU { Pair128 p; v16bf f; };
union PackU { __bf16 h[16]; v16bf f; };

__device__ __forceinline__ float sigm(float v) { return 1.0f / (1.0f + __expf(-v)); }

__device__ __forceinline__ v8f wmma_bf16(v16bf a, v16bf b, v8f c) {
  // (neg_a, A, neg_b, B, c_mod, C, reuse_a, reuse_b) -> v_wmma_f32_16x16x32_bf16
  return __builtin_amdgcn_wmma_f32_16x16x32_bf16(false, a, false, b, (short)0, c,
                                                 false, false);
}

// A-fragment from a bf16 row-major matrix. `base` already points at
// row*ld + K0 + (lane>=16 ? 8 : 0). lane<16 gets K 0..7 & 16..23 of the block.
__device__ __forceinline__ v16bf loadA_bf16(const __bf16* base) {
  FragU u;
  u.p.lo = *(const u32x4*)(base);
  u.p.hi = *(const u32x4*)(base + 16);
  return u.f;
}

// A-fragment from fp32 (the x part of layer0), converted in-register to bf16.
__device__ __forceinline__ v16bf loadA_f32(const float* base) {
  PackU u;
#pragma unroll
  for (int j = 0; j < 8; ++j) u.h[j] = (__bf16)base[j];
#pragma unroll
  for (int j = 0; j < 8; ++j) u.h[8 + j] = (__bf16)base[16 + j];
  return u.f;
}

// B-fragment: weights pre-staged in LDS in exact fragment order; each lane
// reads its 32 contiguous bytes -> two ds_load_b128.
__device__ __forceinline__ v16bf loadB_lds(const char* p) {
  FragU u;
  u.p.lo = *(const u32x4*)(p);
  u.p.hi = *(const u32x4*)(p + 16);
  return u.f;
}

__device__ __forceinline__ void grid_barrier(unsigned* cnt, unsigned* gen,
                                             unsigned target) {
  __threadfence();            // flush this thread's global stores to device scope
  __syncthreads();
  if (threadIdx.x == 0) {
    unsigned prev = atomicAdd(cnt, 1u);
    if (prev == NWG - 1) {
      __hip_atomic_store(cnt, 0u, __ATOMIC_RELAXED, __HIP_MEMORY_SCOPE_AGENT);
      __hip_atomic_fetch_add(gen, 1u, __ATOMIC_RELEASE, __HIP_MEMORY_SCOPE_AGENT);
    } else {
      while (__hip_atomic_load(gen, __ATOMIC_ACQUIRE, __HIP_MEMORY_SCOPE_AGENT) <
             target)
        __builtin_amdgcn_s_sleep(2);
    }
  }
  __syncthreads();
}

__global__ void lstm_init_ws(unsigned* ws) {
  int i = blockIdx.x * blockDim.x + threadIdx.x;
  if (i < WSWORDS) ws[i] = 0u;   // zero h buffers (h(-1)=0) + barrier state
}

__global__ __launch_bounds__(NTHR, 1) void lstm_persistent(
    const float* __restrict__ x, const float* __restrict__ W0,
    const float* __restrict__ b0, const float* __restrict__ W1,
    const float* __restrict__ b1, float* __restrict__ out,
    char* __restrict__ ws) {
  extern __shared__ char smem[];
  const int tid = threadIdx.x;
  const int wg = blockIdx.x;
  const int wgBase = wg * 16;  // first h-column owned by this WG

  __bf16* h0buf = (__bf16*)(ws + H0OFF);
  __bf16* h1buf = (__bf16*)(ws + H1OFF);
  unsigned* cnt = (unsigned*)(ws + BAROFF);
  unsigned* gen = cnt + 1;

  // ---- Stage this WG's weight slice into LDS, bf16, WMMA-B fragment order ----
  // Fragment (tc, kb) = 1KB; lane L's 16 halves: col = L&15, K = kb*32 + (L>=16?8:0) + (j&7) + (j>=8?16:0)
  {
    __bf16* w0l = (__bf16*)(smem + W0LDS);
    for (int e = tid; e < 4 * KB0 * 512; e += NTHR) {
      int frag = e >> 9, q = e & 511;
      int tc = frag / KB0, kb = frag - tc * KB0;
      int ln = q >> 4, elem = q & 15;
      int r = tc * Hh + wgBase + (ln & 15);                 // gate row of W
      int k = kb * 32 + ((ln >> 4) << 3) + (elem & 7) + ((elem >> 3) << 4);  // input column
      w0l[e] = (__bf16)W0[(size_t)r * (Hh + DIN) + k];
    }
    __bf16* w1l = (__bf16*)(smem + W1LDS);
    for (int e = tid; e < 4 * KB1 * 512; e += NTHR) {
      int frag = e >> 9, q = e & 511;
      int tc = frag / KB1, kb = frag - tc * KB1;
      int ln = q >> 4, elem = q & 15;
      int r = tc * Hh + wgBase + (ln & 15);
      int k = kb * 32 + ((ln >> 4) << 3) + (elem & 7) + ((elem >> 3) << 4);
      w1l[e] = (__bf16)W1[(size_t)r * (2 * Hh) + k];
    }
  }
  float* cst = (float*)(smem + CSTO);
  for (int i = tid; i < 2048; i += NTHR) cst[i] = 0.0f;  // c(-1) = 0
  __syncthreads();

  const int wave = tid >> 5, lane = tid & 31;
  const int tc = wave & 3;              // which gate block (i,f,g,o)
  const int mt = (wave >> 2) * 32;      // this wave: output rows mt..mt+31 (2 tiles)
  const int lrow = lane & 15, hsel = lane >> 4;
  const float bias0 = b0[tc * Hh + wgBase + lrow];
  const float bias1 = b1[tc * Hh + wgBase + lrow];
  float* g0 = (float*)(smem + GATESO);
  float* g1 = g0 + 64 * 64;

  // Software pipeline: phase p computes layer0 at t=p and layer1 at t=p-1.
  for (int p = 0; p <= Tt; ++p) {
    const __bf16* h0rd = h0buf + (size_t)((p + 1) & 1) * (Bb * Hh);  // h0(p-1)

    if (p < Tt) {  // -------- layer0 GEMM, step t = p --------
      v8f acc0 = {0.f, 0.f, 0.f, 0.f, 0.f, 0.f, 0.f, 0.f};
      v8f acc1 = acc0;
      const __bf16* ar0 = h0rd + (size_t)(mt + lrow) * Hh + hsel * 8;
      const __bf16* ar1 = h0rd + (size_t)(mt + 16 + lrow) * Hh + hsel * 8;
#pragma unroll 4
      for (int kb = 0; kb < 16; ++kb) {  // recurrent part: K 0..511 (bf16)
        v16bf bf = loadB_lds(smem + W0LDS + ((tc * KB0 + kb) << 10) + lane * 32);
        v16bf a0 = loadA_bf16(ar0 + kb * 32);
        v16bf a1 = loadA_bf16(ar1 + kb * 32);
        acc0 = wmma_bf16(a0, bf, acc0);
        acc1 = wmma_bf16(a1, bf, acc1);
      }
      const float* xr0 = x + ((size_t)(mt + lrow) * Tt + p) * DIN + hsel * 8;
      const float* xr1 = x + ((size_t)(mt + 16 + lrow) * Tt + p) * DIN + hsel * 8;
#pragma unroll 4
      for (int kb = 16; kb < KB0; ++kb) {  // x part: K 512..767 (fp32 -> bf16)
        int c = (kb - 16) * 32;
        v16bf bf = loadB_lds(smem + W0LDS + ((tc * KB0 + kb) << 10) + lane * 32);
        v16bf a0 = loadA_f32(xr0 + c);
        v16bf a1 = loadA_f32(xr1 + c);
        acc0 = wmma_bf16(a0, bf, acc0);
        acc1 = wmma_bf16(a1, bf, acc1);
      }
#pragma unroll
      for (int v = 0; v < 8; ++v) {  // D layout: col=lane&15, row=v+8*(lane>>4)
        g0[(tc * 64 + (mt + 8 * hsel + v)) * 16 + lrow] = acc0[v] + bias0;
        g0[(tc * 64 + (mt + 16 + 8 * hsel + v)) * 16 + lrow] = acc1[v] + bias0;
      }
    }

    if (p >= 1) {  // -------- layer1 GEMM, step t = p-1 --------
      const __bf16* h1rd = h1buf + (size_t)(p & 1) * (Bb * Hh);  // h1(p-2)
      v8f acc0 = {0.f, 0.f, 0.f, 0.f, 0.f, 0.f, 0.f, 0.f};
      v8f acc1 = acc0;
      const __bf16* ar0 = h1rd + (size_t)(mt + lrow) * Hh + hsel * 8;
      const __bf16* ar1 = h1rd + (size_t)(mt + 16 + lrow) * Hh + hsel * 8;
#pragma unroll 4
      for (int kb = 0; kb < 16; ++kb) {  // recurrent part: h1_prev
        v16bf bf = loadB_lds(smem + W1LDS + ((tc * KB1 + kb) << 10) + lane * 32);
        v16bf a0 = loadA_bf16(ar0 + kb * 32);
        v16bf a1 = loadA_bf16(ar1 + kb * 32);
        acc0 = wmma_bf16(a0, bf, acc0);
        acc1 = wmma_bf16(a1, bf, acc1);
      }
      const __bf16* br0 = h0rd + (size_t)(mt + lrow) * Hh + hsel * 8;
      const __bf16* br1 = h0rd + (size_t)(mt + 16 + lrow) * Hh + hsel * 8;
#pragma unroll 4
      for (int kb = 16; kb < KB1; ++kb) {  // input part: h0(p-1)
        int c = (kb - 16) * 32;
        v16bf bf = loadB_lds(smem + W1LDS + ((tc * KB1 + kb) << 10) + lane * 32);
        v16bf a0 = loadA_bf16(br0 + c);
        v16bf a1 = loadA_bf16(br1 + c);
        acc0 = wmma_bf16(a0, bf, acc0);
        acc1 = wmma_bf16(a1, bf, acc1);
      }
#pragma unroll
      for (int v = 0; v < 8; ++v) {
        g1[(tc * 64 + (mt + 8 * hsel + v)) * 16 + lrow] = acc0[v] + bias1;
        g1[(tc * 64 + (mt + 16 + 8 * hsel + v)) * 16 + lrow] = acc1[v] + bias1;
      }
    }

    __syncthreads();  // gates visible WG-wide

    // -------- elementwise cell updates (this WG owns all 4 gates locally) ----
#pragma unroll
    for (int q = 0; q < 4; ++q) {
      int e = tid + (q << 8);           // 0..1023 = row*16 + col
      int r = e >> 4, j = e & 15;
      if (p < Tt) {                     // layer0, step p
        float ig = g0[(0 * 64 + r) * 16 + j];
        float fg = g0[(1 * 64 + r) * 16 + j];
        float gg = g0[(2 * 64 + r) * 16 + j];
        float og = g0[(3 * 64 + r) * 16 + j];
        float c0 = cst[e];
        float cn = sigm(fg) * c0 + sigm(ig) * tanhf(gg);
        float h = sigm(og) * tanhf(cn);
        cst[e] = cn;
        h0buf[(size_t)(p & 1) * (Bb * Hh) + (size_t)r * Hh + wgBase + j] = (__bf16)h;
        if (p == Tt - 1) {
          out[OUT_HN + (size_t)(0 * Bb + r) * Hh + wgBase + j] = h;
          out[OUT_CN + (size_t)(0 * Bb + r) * Hh + wgBase + j] = cn;
        }
      }
      if (p >= 1) {                     // layer1, step p-1
        float ig = g1[(0 * 64 + r) * 16 + j];
        float fg = g1[(1 * 64 + r) * 16 + j];
        float gg = g1[(2 * 64 + r) * 16 + j];
        float og = g1[(3 * 64 + r) * 16 + j];
        float c1 = cst[1024 + e];
        float cn = sigm(fg) * c1 + sigm(ig) * tanhf(gg);
        float h = sigm(og) * tanhf(cn);
        cst[1024 + e] = cn;
        h1buf[(size_t)((p - 1) & 1) * (Bb * Hh) + (size_t)r * Hh + wgBase + j] =
            (__bf16)h;
        out[((size_t)r * Tt + (p - 1)) * Hh + wgBase + j] = h;  // outputs (B,T,H)
        if (p == Tt) {
          out[OUT_HN + (size_t)(1 * Bb + r) * Hh + wgBase + j] = h;
          out[OUT_CN + (size_t)(1 * Bb + r) * Hh + wgBase + j] = cn;
        }
      }
    }

    // Prefetch next timestep's x rows into cache (global_prefetch_b8).
    if (p + 1 < Tt) {
      int r = tid & 63;
      __builtin_prefetch(x + ((size_t)r * Tt + (p + 1)) * DIN + ((tid >> 6) * 64),
                         0, 0);
    }

    grid_barrier(cnt, gen, (unsigned)(p + 1));
  }
}

extern "C" void kernel_launch(void* const* d_in, const int* in_sizes, int n_in,
                              void* d_out, int out_size, void* d_ws, size_t ws_size,
                              hipStream_t stream) {
  const float* x = (const float*)d_in[0];
  const float* W0 = (const float*)d_in[1];
  const float* b0 = (const float*)d_in[2];
  const float* W1 = (const float*)d_in[3];
  const float* b1 = (const float*)d_in[4];
  float* out = (float*)d_out;
  char* ws = (char*)d_ws;

  (void)hipFuncSetAttribute((const void*)lstm_persistent,
                            hipFuncAttributeMaxDynamicSharedMemorySize, LDS_TOTAL);

  lstm_init_ws<<<(WSWORDS + 255) / 256, 256, 0, stream>>>((unsigned*)ws);
  lstm_persistent<<<dim3(NWG), dim3(NTHR), LDS_TOTAL, stream>>>(x, W0, b0, W1, b1,
                                                                out, ws);
}
